// Set2Set_50714973831822
// MI455X (gfx1250) — matrix operand order repaired and checked by reading, not compiled, not run on hardware
//
#include <hip/hip_runtime.h>

// Problem constants (from reference): N=262144 nodes, D=512, B=4096 graphs, 3 steps.
#define N_NODES 262144
#define DIM     512
#define NBATCH  4096
#define NSTEPS  3
#define KTOT    1536   // 2D (q_star) + D (h)  -> single fused K for the LSTM GEMM
#define NGATE   2048   // 4*D gate outputs

typedef __attribute__((ext_vector_type(16))) __bf16 v16bf;
typedef __attribute__((ext_vector_type(8)))  __bf16 v8bf;
typedef __attribute__((ext_vector_type(8)))  float  v8f;

__device__ __forceinline__ float sigmoidf_(float v) { return 1.0f / (1.0f + expf(-v)); }

// ---------------------------------------------------------------------------
// Pack [W_ih | W_hh] into bf16 Wcat[2048][1536] (row n = all weights feeding
// gate-output n, contiguous in K -> B-fragment loads are contiguous per lane).
// ---------------------------------------------------------------------------
__global__ void pack_weights_kernel(const float* __restrict__ W_ih,
                                    const float* __restrict__ W_hh,
                                    __bf16* __restrict__ Wcat) {
  int t = blockIdx.x * blockDim.x + threadIdx.x;      // exactly 2048*1536 threads
  int n = t / KTOT;
  int k = t - n * KTOT;
  float v = (k < 1024) ? W_ih[(size_t)n * 1024 + k]
                       : W_hh[(size_t)n * 512 + (k - 1024)];
  Wcat[(size_t)n * KTOT + k] = (__bf16)v;
}

__global__ void zero_kernel(float4* __restrict__ p, int n4) {
  int t = blockIdx.x * blockDim.x + threadIdx.x;
  if (t < n4) p[t] = make_float4(0.f, 0.f, 0.f, 0.f);
}

// ---------------------------------------------------------------------------
// Fused LSTM step. Grid (256,32): blockIdx.x = 16-row batch tile,
// blockIdx.y = 16-col tile inside each gate. 4 waves/block, wave g computes
// gate g's 16x16 tile with v_wmma_f32_16x16x32_bf16 over K=1536, then the
// i/f/g/o tiles meet in LDS for the fused elementwise LSTM update.
// A layout: [4096][1536] bf16 = [h | r | h] from the previous step.
// ---------------------------------------------------------------------------
__global__ __launch_bounds__(128) void lstm_gemm_kernel(
    const __bf16* __restrict__ A,      // [4096][1536] bf16 activations (prev step)
    const __bf16* __restrict__ Wcat,   // [2048][1536] bf16 weights
    const float*  __restrict__ b_ih,
    const float*  __restrict__ b_hh,
    float*        __restrict__ c,      // [4096][512] cell state (in/out)
    float*        __restrict__ qstar,  // [4096][1024] f32; cols [0,512) <- h
    __bf16*       __restrict__ A_next) // [4096][1536] bf16; h -> cols [0,512)+[1024,1536)
{
  __shared__ float gbuf[4][16][16];
  const int bm   = blockIdx.x;        // batch-row tile
  const int bn   = blockIdx.y;        // column tile within each gate
  const int lane = threadIdx.x & 31;
  const int g    = threadIdx.x >> 5;  // gate id: 0=i 1=f 2=g 3=o (PyTorch order)
  const int lo   = lane & 15;
  const int hi   = lane >> 4;

  // A fragment (16x32 bf16): lanes 0-15 row m=lo hold K {0..15},
  // lanes 16-31 hold K {16..31} -> two contiguous 16B chunks per lane.
  const __bf16* arow = A    + (size_t)(bm * 16 + lo) * KTOT + hi * 16;
  // B fragment (32x16 bf16): lane holds a contiguous 16-K run of column n=lo
  // (lanes 16-31 take the upper 16 K) -> one 32B contiguous load from Wcat row.
  const __bf16* wrow = Wcat + (size_t)(g * 512 + bn * 16 + lo) * KTOT + hi * 16;

  v8f acc = {0.f, 0.f, 0.f, 0.f, 0.f, 0.f, 0.f, 0.f};
  for (int k0 = 0; k0 < KTOT; k0 += 32) {
    v8bf a0 = *(const v8bf*)(arow + k0);
    v8bf a1 = *(const v8bf*)(arow + k0 + 8);
    v16bf af = __builtin_shufflevector(a0, a1,
        0, 1, 2, 3, 4, 5, 6, 7, 8, 9, 10, 11, 12, 13, 14, 15);
    v16bf bf = *(const v16bf*)(wrow + k0);
    acc = __builtin_amdgcn_wmma_f32_16x16x32_bf16(
        false, af, false, bf, (short)0, acc, false, false);
  }

  // C tile layout: VGPR e on lane l holds element (m = e + 8*(l>>4), n = l&15).
  const int ncol  = g * 512 + bn * 16 + lo;
  const float bias = b_ih[ncol] + b_hh[ncol];
  #pragma unroll
  for (int e = 0; e < 8; ++e)
    gbuf[g][e + 8 * hi][lo] = acc[e] + bias;
  __syncthreads();

  // Fused LSTM elementwise update: 256 elements, 128 threads -> 2 each.
  for (int e = threadIdx.x; e < 256; e += 128) {
    int m = e >> 4, n = e & 15;
    float gi = sigmoidf_(gbuf[0][m][n]);
    float gf = sigmoidf_(gbuf[1][m][n]);
    float gg = tanhf    (gbuf[2][m][n]);
    float go = sigmoidf_(gbuf[3][m][n]);
    int row = bm * 16 + m;
    int col = bn * 16 + n;
    size_t cidx = (size_t)row * DIM + col;
    float cn = gf * c[cidx] + gi * gg;
    float h  = go * tanhf(cn);
    c[cidx] = cn;
    qstar[(size_t)row * 1024 + col] = h;           // q = h (f32, read by attention)
    __bf16 hb = (__bf16)h;
    A_next[(size_t)row * KTOT + col]        = hb;  // q_star[:, :512]
    A_next[(size_t)row * KTOT + 1024 + col] = hb;  // h operand of W_hh
  }
}

// ---------------------------------------------------------------------------
// Single-pass attention: attn = x.q[batch]; accumulate exp-sum (den) and
// un-normalized weighted sum (r_raw) per segment. batch is sorted, so each
// wave walks 8 consecutive nodes holding the full 512-wide accumulator in
// registers (16 f32/lane) and flushes with f32 atomics only on segment change.
// ---------------------------------------------------------------------------
__device__ __forceinline__ void flush_seg(float* __restrict__ r_raw,
                                          float* __restrict__ den,
                                          int seg, int lane,
                                          float* racc, float& dacc) {
  float* rb = r_raw + (size_t)seg * DIM;
  #pragma unroll
  for (int ch = 0; ch < 4; ++ch)
    #pragma unroll
    for (int j = 0; j < 4; ++j)
      atomicAdd(rb + ch * 128 + lane * 4 + j, racc[ch * 4 + j]);
  if (lane == 0) atomicAdd(den + seg, dacc);
  #pragma unroll
  for (int j = 0; j < 16; ++j) racc[j] = 0.f;
  dacc = 0.f;
}

__global__ __launch_bounds__(256) void attn_kernel(
    const float* __restrict__ x,       // [262144][512]
    const int*   __restrict__ batch,   // [262144], sorted
    const float* __restrict__ qstar,   // q = cols [0,512) of [4096][1024]
    float*       __restrict__ r_raw,   // [4096][512] accum (zeroed)
    float*       __restrict__ den)     // [4096] accum (zeroed)
{
  const int lane = threadIdx.x & 31;
  const int wave = threadIdx.x >> 5;
  const int base = blockIdx.x * 64 + wave * 8;   // 8 consecutive nodes per wave

  float racc[16];
  #pragma unroll
  for (int j = 0; j < 16; ++j) racc[j] = 0.f;
  float dacc = 0.f;
  int cur = batch[base];

  for (int t = 0; t < 8; ++t) {
    int nid = base + t;
    int seg = batch[nid];
    if (seg != cur) {
      flush_seg(r_raw, den, cur, lane, racc, dacc);
      cur = seg;
    }
    const float* xr = x + (size_t)nid * DIM;
    const float* qr = qstar + (size_t)seg * 1024;
    if (t < 7) __builtin_prefetch(x + (size_t)(nid + 1) * DIM + lane * 4, 0, 0);

    float4 xv[4];
    float dot = 0.f;
    #pragma unroll
    for (int ch = 0; ch < 4; ++ch) {
      xv[ch]    = *(const float4*)(xr + ch * 128 + lane * 4);
      float4 qv = *(const float4*)(qr + ch * 128 + lane * 4);
      dot += xv[ch].x * qv.x + xv[ch].y * qv.y + xv[ch].z * qv.z + xv[ch].w * qv.w;
    }
    #pragma unroll
    for (int m = 16; m >= 1; m >>= 1) dot += __shfl_xor(dot, m, 32);

    float e = expf(dot);           // reference applies exp without max-shift
    dacc += e;
    #pragma unroll
    for (int ch = 0; ch < 4; ++ch) {
      racc[ch * 4 + 0] += e * xv[ch].x;
      racc[ch * 4 + 1] += e * xv[ch].y;
      racc[ch * 4 + 2] += e * xv[ch].z;
      racc[ch * 4 + 3] += e * xv[ch].w;
    }
  }
  flush_seg(r_raw, den, cur, lane, racc, dacc);
}

// r = r_raw / (den + 1e-8); write into q_star[:, 512:] (f32) and next A (bf16).
__global__ void finalize_kernel(const float* __restrict__ r_raw,
                                const float* __restrict__ den,
                                float*  __restrict__ qstar,
                                __bf16* __restrict__ A_next) {
  int t = blockIdx.x * blockDim.x + threadIdx.x;   // exactly 4096*512 threads
  int row = t >> 9;
  int col = t & 511;
  float r = r_raw[t] / (den[row] + 1e-8f);
  qstar[(size_t)row * 1024 + 512 + col]  = r;
  A_next[(size_t)row * KTOT + 512 + col] = (__bf16)r;
}

// ---------------------------------------------------------------------------
// Workspace layout (bytes, all 256B-aligned):
//   Wcat   bf16 [2048][1536] :  6,291,456
//   A0     bf16 [4096][1536] : 12,582,912
//   A1     bf16 [4096][1536] : 12,582,912
//   qstar  f32  [4096][1024] : 16,777,216
//   c      f32  [4096][512]  :  8,388,608
//   r_raw  f32  [4096][512]  :  8,388,608
//   den    f32  [4096]       :     16,384
// total ~65 MB
// ---------------------------------------------------------------------------
extern "C" void kernel_launch(void* const* d_in, const int* in_sizes, int n_in,
                              void* d_out, int out_size, void* d_ws, size_t ws_size,
                              hipStream_t stream) {
  const float* x     = (const float*)d_in[0];
  const int*   batch = (const int*)  d_in[1];
  const float* W_ih  = (const float*)d_in[2];
  const float* W_hh  = (const float*)d_in[3];
  const float* b_ih  = (const float*)d_in[4];
  const float* b_hh  = (const float*)d_in[5];

  char* ws = (char*)d_ws;
  size_t off = 0;
  __bf16* Wcat  = (__bf16*)(ws + off); off += (size_t)NGATE * KTOT * 2;       //  6,291,456
  __bf16* A0    = (__bf16*)(ws + off); off += (size_t)NBATCH * KTOT * 2;      // 12,582,912
  __bf16* A1    = (__bf16*)(ws + off); off += (size_t)NBATCH * KTOT * 2;      // 12,582,912
  float*  qstar = (float*) (ws + off); off += (size_t)NBATCH * 1024 * 4;      // 16,777,216
  float*  cbuf  = (float*) (ws + off); off += (size_t)NBATCH * DIM * 4;       //  8,388,608
  float*  r_raw = (float*) (ws + off); off += (size_t)NBATCH * DIM * 4;       //  8,388,608
  float*  den   = (float*) (ws + off); off += (size_t)NBATCH * 4;             //     16,384

  // Pack weights to bf16 (2048*1536 / 256 = 12288 blocks).
  pack_weights_kernel<<<12288, 256, 0, stream>>>(W_ih, W_hh, Wcat);

  // Zero initial activations A0 (q_star = h = 0) and cell state c.
  zero_kernel<<<((size_t)NBATCH * KTOT * 2 / 16 + 255) / 256, 256, 0, stream>>>(
      (float4*)A0, (int)((size_t)NBATCH * KTOT * 2 / 16));
  zero_kernel<<<((size_t)NBATCH * DIM * 4 / 16 + 255) / 256, 256, 0, stream>>>(
      (float4*)cbuf, (int)((size_t)NBATCH * DIM * 4 / 16));

  __bf16* Acur = A0;
  __bf16* Anxt = A1;
  for (int step = 0; step < NSTEPS; ++step) {
    // LSTM cell (WMMA bf16 GEMM + fused epilogue) -> h, c, next-A cols [0,512)+[1024,1536).
    lstm_gemm_kernel<<<dim3(NBATCH / 16, DIM / 16), 128, 0, stream>>>(
        Acur, Wcat, b_ih, b_hh, cbuf, qstar, Anxt);

    // Zero r_raw + den (contiguous region).
    size_t zb = ((size_t)NBATCH * DIM * 4 + (size_t)NBATCH * 4) / 16;
    zero_kernel<<<(zb + 255) / 256, 256, 0, stream>>>((float4*)r_raw, (int)zb);

    // One streaming pass over x: dot, exp, segmented accumulation.
    attn_kernel<<<N_NODES / 64, 256, 0, stream>>>(x, batch, qstar, r_raw, den);

    // Normalize -> r into q_star[:,512:] and next-A cols [512,1024).
    finalize_kernel<<<(NBATCH * DIM) / 256, 256, 0, stream>>>(r_raw, den, qstar, Anxt);

    __bf16* tmp = Acur; Acur = Anxt; Anxt = tmp;
  }

  // q_star after step 3 is the output [4096][1024] f32.
  hipMemcpyAsync(d_out, qstar, (size_t)NBATCH * 1024 * sizeof(float),
                 hipMemcpyDeviceToDevice, stream);
}
